// RankLSTM_64647847739678
// MI455X (gfx1250) — matrix operand order repaired and checked by reading, not compiled, not run on hardware
//
#include <hip/hip_runtime.h>
#include <hip/hip_bf16.h>

// ---------------------------------------------------------------------------
// RankLSTM for MI455X (gfx1250, wave32).
//   Inputs (setup_inputs order):
//     0 feature [4096,512,5] f32   1 ret [4096] f32   2 mask [4096] bool(u8)
//     3 Wih [256,5] f32  4 Whh [256,64] f32  5 bih [256]  6 bhh [256]
//     7 Wd [1,64]        8 bd [1]
//   Output: pred[4096] | loss | reg_loss | rank_loss  (4099 f32)
//   Workspace: h_last [4096*64] f32 + 4 accumulator floats.
// ---------------------------------------------------------------------------

#define HID  64
#define GATES 256          // 4*HID
#define NF   5
#define TSTEPS 512
#define NSEQ 4096
#define NEG_SLOPE 0.2f

#define ROWS 16            // sequences per workgroup
#define CT   32            // timesteps per staged LDS chunk (double buffered)
#define HPAD 66            // padded h row stride (bank-conflict free, even)
#define GPAD 18            // padded gates column stride (even)
#define XW   (CT * NF)     // 160 floats per row per chunk

typedef __attribute__((ext_vector_type(16))) __bf16 v16bf;
typedef __attribute__((ext_vector_type(8)))  float  v8f;
typedef __attribute__((ext_vector_type(4)))  int    v4i;

union V16 { v16bf v; unsigned short s[16]; };
union A8  { v8f   v; float f[8]; };

#if __has_builtin(__builtin_amdgcn_global_load_async_to_lds_b128)
#define HAVE_ASYNC_LDS 1
#else
#define HAVE_ASYNC_LDS 0
#endif

#if HAVE_ASYNC_LDS
// 16-byte async DMA global -> LDS (ASYNCcnt-tracked, no VGPR round trip).
// Param types per hipcc diagnostic: (as1 v4i*, as3 v4i*, i32 offset, i32 cpol)
__device__ __forceinline__ void async_copy16(const float* g, float* l) {
  __builtin_amdgcn_global_load_async_to_lds_b128(
      (__attribute__((address_space(1))) v4i*)g,
      (__attribute__((address_space(3))) v4i*)l,
      0, 0);
}
#endif

__device__ __forceinline__ void async_wait_all() {
#if HAVE_ASYNC_LDS
#if __has_builtin(__builtin_amdgcn_s_wait_asynccnt)
  __builtin_amdgcn_s_wait_asynccnt(0);
#else
  asm volatile("s_wait_asynccnt 0x0" ::: "memory");
#endif
#endif
}

__device__ __forceinline__ unsigned short f2bf(float x) {
  union { float f; unsigned u; } c; c.f = x;
  unsigned u = c.u + (0x7FFFu + ((c.u >> 16) & 1u));   // round-to-nearest-even
  return (unsigned short)(u >> 16);
}

// branch-free sigmoid: v_exp_f32 + v_rcp_f32 (TRANS ops, co-execute with WMMA)
__device__ __forceinline__ float sigmoidf_fast(float x) {
  return __builtin_amdgcn_rcpf(1.f + __expf(-x));
}

// branch-free tanh: sign(x) * (1 - e^-2|x|) / (1 + e^-2|x|), rcp instead of div
__device__ __forceinline__ float tanhf_fast(float x) {
  const float e = __expf(-2.f * __builtin_fabsf(x));
  const float t = (1.f - e) * __builtin_amdgcn_rcpf(1.f + e);
  return __builtin_copysignf(t, x);
}

// ---------------------------------------------------------------------------
// LSTM: one workgroup = 16 sequences, 4 waves, wave w owns gate group w.
// Per step per wave: 4 output tiles 16x16, each = 2 chained bf16 WMMAs (K=64),
// C operand seeded with bias + x_t @ Wih^T (VALU, co-executes with WMMA).
// Feature chunks are DMA'd into LDS with GLOBAL_LOAD_ASYNC_TO_LDS (double
// buffered) when the toolchain exposes the builtin; sync-copy fallback else.
// ---------------------------------------------------------------------------
__global__ __launch_bounds__(128)
void rl_lstm_kernel(const float* __restrict__ feat,
                    const float* __restrict__ Wih,
                    const float* __restrict__ Whh,
                    const float* __restrict__ bih,
                    const float* __restrict__ bhh,
                    float* __restrict__ hlast)
{
  __shared__ float hbuf[ROWS * HPAD];     // h state, fp32
  __shared__ float gbuf[GATES * GPAD];    // gate pre-activations [col][row]
  __shared__ float xbuf[2][ROWS * XW];    // double-buffered feature chunks

  const int tid   = threadIdx.x;
  const int wave  = tid >> 5;             // 0..3 -> gates i,f,g,o
  const int lane  = tid & 31;
  const int col   = lane & 15;            // output column within 16x16 tile
  const int rbase = (lane < 16) ? 0 : 8;  // C-layout row base for this lane
  const int n0    = blockIdx.x * ROWS;

  // stage one chunk of features (ROWS x CT x NF) into an LDS buffer
  auto stage_chunk = [&](int tbase, float* dst) {
    for (int i = tid; i < ROWS * (XW / 4); i += 128) {
      const int row = i / (XW / 4);
      const int q   = i - row * (XW / 4);
      const float* g = feat + ((size_t)(n0 + row) * TSTEPS + tbase) * NF + 4 * q;
      float* l = dst + row * XW + 4 * q;
#if HAVE_ASYNC_LDS
      async_copy16(g, l);
#else
      *(float4*)l = *(const float4*)g;
#endif
    }
  };

  // ---- loop-invariant: Whh^T as bf16 B operands, Wih/bias per lane --------
  V16   Bm[4][2];
  float wih_r[4][NF];
  float bias_r[4];
  #pragma unroll
  for (int tt = 0; tt < 4; ++tt) {
    const int gc = wave * 64 + tt * 16 + col;      // gate column 0..255
    #pragma unroll
    for (int kb = 0; kb < 2; ++kb) {
      const int kbase = kb * 32 + ((lane < 16) ? 0 : 16);
      #pragma unroll
      for (int v = 0; v < 8; ++v) {
        const int k0 = kbase + 2 * v;              // B layout: 2 K per VGPR
        Bm[tt][kb].s[2 * v]     = f2bf(Whh[gc * HID + k0]);
        Bm[tt][kb].s[2 * v + 1] = f2bf(Whh[gc * HID + k0 + 1]);
      }
    }
    #pragma unroll
    for (int f = 0; f < NF; ++f) wih_r[tt][f] = Wih[gc * NF + f];
    bias_r[tt] = bih[gc] + bhh[gc];
  }

  // ---- state init + prefetch chunk 0 --------------------------------------
  stage_chunk(0, xbuf[0]);
  for (int i = tid; i < ROWS * HPAD; i += 128) hbuf[i] = 0.f;
  float c_reg[8];
  #pragma unroll
  for (int r = 0; r < 8; ++r) c_reg[r] = 0.f;
  const int jj     = tid & 63;                  // hidden index owned in cell update
  const int nbase  = (tid < 64) ? 0 : 8;        // rows owned in cell update

  for (int t = 0; t < TSTEPS; ++t) {
    const int lt    = t & (CT - 1);
    const int chunk = t >> 5;                   // CT == 32
    if (lt == 0) {
      async_wait_all();                         // this chunk's DMA complete
      __syncthreads();                          // ...and visible to all waves
      if (t + CT < TSTEPS)                      // kick off next chunk's DMA
        stage_chunk(t + CT, xbuf[(chunk + 1) & 1]);
    }
    const float* xb = xbuf[chunk & 1];

    // ---- build bf16 A operands from h (contiguous k order k = 16*(lane/16) + s, A1 at +32, K=0..63) ----
    V16 A0, A1;
    {
      const int m = col;
      #pragma unroll
      for (int v = 0; v < 8; ++v) {
        const int k0 = 2 * v + ((lane < 16) ? 0 : 16);
        const float2 h0 = *(const float2*)(hbuf + m * HPAD + k0);
        const float2 h1 = *(const float2*)(hbuf + m * HPAD + k0 + 32);
        A0.s[2 * v] = f2bf(h0.x); A0.s[2 * v + 1] = f2bf(h0.y);
        A1.s[2 * v] = f2bf(h1.x); A1.s[2 * v + 1] = f2bf(h1.y);
      }
    }

    // per-lane x_t values for the 8 C rows (broadcast LDS reads)
    float xr[8][NF];
    #pragma unroll
    for (int r = 0; r < 8; ++r)
      #pragma unroll
      for (int f = 0; f < NF; ++f)
        xr[r][f] = xb[(r + rbase) * XW + lt * NF + f];

    // ---- gates = (bias + x_t@Wih^T) + h@Whh^T, 4 tiles per wave ----------
    #pragma unroll
    for (int tt = 0; tt < 4; ++tt) {
      A8 acc;
      #pragma unroll
      for (int r = 0; r < 8; ++r) {
        float s = bias_r[tt];
        #pragma unroll
        for (int f = 0; f < NF; ++f) s = fmaf(xr[r][f], wih_r[tt][f], s);
        acc.f[r] = s;
      }
      acc.v = __builtin_amdgcn_wmma_f32_16x16x32_bf16(
                  false, A0.v, false, Bm[tt][0].v, (short)0, acc.v, false, false);
      acc.v = __builtin_amdgcn_wmma_f32_16x16x32_bf16(
                  false, A1.v, false, Bm[tt][1].v, (short)0, acc.v, false, false);
      const int gc = wave * 64 + tt * 16 + col;
      float* gp = gbuf + gc * GPAD + rbase;
      #pragma unroll
      for (int r = 0; r < 8; r += 2)
        *(float2*)(gp + r) = make_float2(acc.f[r], acc.f[r + 1]);
    }
    __syncthreads();

    // ---- cell update: thread owns (rows nbase..nbase+7, hidden jj) -------
    #pragma unroll
    for (int r = 0; r < 8; ++r) {
      const int n = nbase + r;
      const float gi = gbuf[(jj)       * GPAD + n];
      const float gf = gbuf[(64 + jj)  * GPAD + n];
      const float gg = gbuf[(128 + jj) * GPAD + n];
      const float go = gbuf[(192 + jj) * GPAD + n];
      const float i = sigmoidf_fast(gi);
      const float f = sigmoidf_fast(gf);
      const float g = tanhf_fast(gg);
      const float o = sigmoidf_fast(go);
      const float c = f * c_reg[r] + i * g;
      c_reg[r] = c;
      hbuf[n * HPAD + jj] = o * tanhf_fast(c);
    }
    __syncthreads();
  }

  // ---- emit last hidden state --------------------------------------------
  for (int i = tid; i < ROWS * HID; i += 128) {
    const int n = i >> 6, j = i & 63;
    hlast[(size_t)(n0 + n) * HID + j] = hbuf[n * HPAD + j];
  }
}

// ---------------------------------------------------------------------------
// pred = leaky_relu(h_last @ Wd^T + bd)
// ---------------------------------------------------------------------------
__global__ __launch_bounds__(256)
void rl_head_kernel(const float* __restrict__ hlast,
                    const float* __restrict__ Wd,
                    const float* __restrict__ bd,
                    float* __restrict__ pred)
{
  __shared__ float w[HID];
  if (threadIdx.x < HID) w[threadIdx.x] = Wd[threadIdx.x];
  __syncthreads();
  const int n = blockIdx.x * 256 + threadIdx.x;
  if (n < NSEQ) {
    float s = bd[0];
    const float* hp = hlast + (size_t)n * HID;
    #pragma unroll 8
    for (int j = 0; j < HID; ++j) s = fmaf(hp[j], w[j], s);
    pred[n] = (s > 0.f) ? s : NEG_SLOPE * s;
  }
}

__global__ void rl_init_kernel(float* acc) {
  if (threadIdx.x < 4) acc[threadIdx.x] = 0.f;
}

// ---------------------------------------------------------------------------
// Masked MSE partials + O(N^2) rank loss with LDS-tiled j loop.
// acc[0]=sum (pred-ret)^2*m, acc[1]=sum m, acc[2]=sum relu(-dp*dt*mm)
// ---------------------------------------------------------------------------
__global__ __launch_bounds__(256)
void rl_loss_kernel(const float* __restrict__ pred,
                    const float* __restrict__ ret,
                    const unsigned char* __restrict__ mask,
                    float* __restrict__ acc)
{
  __shared__ float pj[256], rj[256], mj[256], red[256];
  const int i  = blockIdx.x * 256 + threadIdx.x;
  const float pi = pred[i];
  const float ri = ret[i];
  const float mi = mask[i] ? 1.f : 0.f;

  float rank = 0.f;
  for (int j0 = 0; j0 < NSEQ; j0 += 256) {
    __syncthreads();
    pj[threadIdx.x] = pred[j0 + threadIdx.x];
    rj[threadIdx.x] = ret[j0 + threadIdx.x];
    mj[threadIdx.x] = mask[j0 + threadIdx.x] ? 1.f : 0.f;
    __syncthreads();
    #pragma unroll 8
    for (int j = 0; j < 256; ++j) {
      const float v = -(pj[j] - pi) * (rj[j] - ri) * (mi * mj[j]);
      rank += (v > 0.f) ? v : 0.f;
    }
  }

  float vals[3] = { (pi - ri) * (pi - ri) * mi, mi, rank };
  #pragma unroll
  for (int k = 0; k < 3; ++k) {
    __syncthreads();
    red[threadIdx.x] = vals[k];
    __syncthreads();
    for (int s = 128; s > 0; s >>= 1) {
      if (threadIdx.x < s) red[threadIdx.x] += red[threadIdx.x + s];
      __syncthreads();
    }
    if (threadIdx.x == 0) atomicAdd(acc + k, red[0]);
  }
}

__global__ void rl_final_kernel(const float* __restrict__ acc,
                                float* __restrict__ out)
{
  const float reg  = acc[0] / (acc[1] + 1e-8f);
  const float rank = acc[2] / ((float)NSEQ * (float)NSEQ);
  out[0] = reg + rank;   // loss
  out[1] = reg;          // reg_loss
  out[2] = rank;         // rank_loss
}

// ---------------------------------------------------------------------------
extern "C" void kernel_launch(void* const* d_in, const int* in_sizes, int n_in,
                              void* d_out, int out_size, void* d_ws, size_t ws_size,
                              hipStream_t stream) {
  const float*         feat = (const float*)d_in[0];
  const float*         ret  = (const float*)d_in[1];
  const unsigned char* mask = (const unsigned char*)d_in[2];  // jax bool -> 1 byte
  const float*         Wih  = (const float*)d_in[3];
  const float*         Whh  = (const float*)d_in[4];
  const float*         bih  = (const float*)d_in[5];
  const float*         bhh  = (const float*)d_in[6];
  const float*         Wd   = (const float*)d_in[7];
  const float*         bd   = (const float*)d_in[8];

  float* out   = (float*)d_out;                 // pred[4096] | loss | reg | rank
  float* hlast = (float*)d_ws;                  // 4096*64 f32
  float* acc   = hlast + (size_t)NSEQ * HID;    // 4 accumulators

  rl_lstm_kernel<<<NSEQ / ROWS, 128, 0, stream>>>(feat, Wih, Whh, bih, bhh, hlast);
  rl_head_kernel<<<NSEQ / 256, 256, 0, stream>>>(hlast, Wd, bd, out);
  rl_init_kernel<<<1, 32, 0, stream>>>(acc);
  rl_loss_kernel<<<NSEQ / 256, 256, 0, stream>>>(out, ret, mask, acc);
  rl_final_kernel<<<1, 1, 0, stream>>>(acc, out + NSEQ);
}
